// Attention_38019050504723
// MI455X (gfx1250) — compile-verified, matrix operand order checked by reading, not run on hardware
//
#include <hip/hip_runtime.h>
#include <hip/hip_bf16.h>
#include <math.h>

typedef __attribute__((ext_vector_type(16))) _Float16 v16h;
typedef __attribute__((ext_vector_type(8)))  float    v8f;
typedef __attribute__((ext_vector_type(4)))  unsigned int v4u;
typedef __attribute__((ext_vector_type(8)))  int      v8i;
typedef __attribute__((ext_vector_type(4)))  int      v4i;

#define BB 8
#define SS 1024
#define EE 256
#define HH 8
#define DD 32
#define NORMF 0.17677669529663687f   // 32^-0.5

// ---------- WMMA helper ----------
__device__ __forceinline__ v8f wmma16(v16h a, v16h b, v8f c) {
  return __builtin_amdgcn_wmma_f32_16x16x32_f16(false, a, false, b, (short)0, c,
                                                false, false);
}

// ---------- A/B operand loaders ----------
// 16-bit A-matrix 16x32 layout (ISA 7.12.2): lane L holds row r = L%16,
// halves 0..7 -> K = g*8 + 0..7, halves 8..15 -> K = g*8 + 16..23 (g = L/16).
// B (KxN) mirrors this with lane = column.
__device__ __forceinline__ v16h load_a16_f32(const float* base, int ld) {
  const int lane = threadIdx.x & 31;
  const int r = lane & 15, g = lane >> 4;
  const float* p = base + (size_t)r * ld + g * 8;
  v16h o;
#pragma unroll
  for (int i = 0; i < 8; ++i) {
    o[i]     = (_Float16)p[i];
    o[i + 8] = (_Float16)p[i + 16];
  }
  return o;
}
__device__ __forceinline__ v16h load_a16_f16(const _Float16* base, int ld) {
  const int lane = threadIdx.x & 31;
  const int r = lane & 15, g = lane >> 4;
  const _Float16* p = base + (size_t)r * ld + g * 8;
  v16h o;
#pragma unroll
  for (int i = 0; i < 8; ++i) {
    o[i]     = p[i];
    o[i + 8] = p[i + 16];
  }
  return o;
}

// ---------- half-wave (16-lane) row reductions (wave32) ----------
__device__ __forceinline__ float hmax16(float v) {
#pragma unroll
  for (int m = 8; m; m >>= 1) v = fmaxf(v, __shfl_xor(v, m, 32));
  return v;
}
__device__ __forceinline__ float hsum16(float v) {
#pragma unroll
  for (int m = 8; m; m >>= 1) v += __shfl_xor(v, m, 32);
  return v;
}

// ---------- Tensor Data Mover: DMA a 16x32 fp32 tile (row stride SS) ----------
// Descriptor per CDNA5 ISA ch.8: group0 {count, lds_addr, global_addr, type=2},
// group1 {data_size=4B, tensor_dims=SS, tile=32x16, dim0_stride=SS}.
__device__ __forceinline__ void tdm_load_bias_tile(const float* gptr,
                                                   unsigned lds_off) {
  const unsigned long long ga = (unsigned long long)(uintptr_t)gptr; // byte addr
  v4u g0;
  g0[0] = 1u;                                    // count=1 (valid), user mode
  g0[1] = lds_off;                               // lds_addr (bytes)
  g0[2] = (unsigned)(ga & 0xffffffffu);          // global_addr[31:0]
  g0[3] = (unsigned)((ga >> 32) & 0x01ffffffu)   // global_addr[56:32]
          | (2u << 30);                          // type = 2 ("image")
  v8i g1;
  g1[0] = 0x00020000;                            // data_size=2 (4B), mask/pad=0
  g1[1] = (int)((SS & 0xffffu) << 16);           // tensor_dim0[15:0] @ bits63:48
  g1[2] = (int)(((SS >> 16) & 0xffffu) | ((SS & 0xffffu) << 16)); // d0 hi | d1 lo
  g1[3] = (int)(((SS >> 16) & 0xffffu) | (32u << 16));            // d1 hi | tile_dim0=32
  g1[4] = 16;                                    // tile_dim1=16 rows, tile_dim2=0
  g1[5] = SS;                                    // tensor_dim0_stride[31:0]
  g1[6] = 0;                                     // stride hi / dim1_stride lo
  g1[7] = 0;
  v4i zg = {0, 0, 0, 0};
#if defined(__clang_major__) && (__clang_major__ >= 23)
  v8i zg8 = {};
  __builtin_amdgcn_tensor_load_to_lds(g0, g1, zg, zg, zg8, 0);
#else
  __builtin_amdgcn_tensor_load_to_lds(g0, g1, zg, zg, 0);
#endif
}

// =====================================================================
// Kernel 1: fused projections.  z = 0: qh (scaled)  1: gate  2: kh  3: vhT
// One wave computes one 16x16 output tile, K=256 -> 8 WMMAs.
// =====================================================================
__global__ __launch_bounds__(32)
void proj_kernel(const float* __restrict__ q, const float* __restrict__ k,
                 const float* __restrict__ v,
                 const float* __restrict__ Wq, const float* __restrict__ Wk,
                 const float* __restrict__ Wv, const float* __restrict__ Wg,
                 const float* __restrict__ bg,
                 _Float16* __restrict__ qh, _Float16* __restrict__ kh,
                 _Float16* __restrict__ vhT, float* __restrict__ gate) {
  const int r0 = blockIdx.x * 16;  // row in flattened [B*S, E]
  const int c0 = blockIdx.y * 16;  // output feature tile
  const int z  = blockIdx.z;
  const float* X = (z <= 1) ? q : (z == 2 ? k : v);
  const float* W = (z == 0) ? Wq : (z == 1 ? Wg : (z == 2 ? Wk : Wv));

  v8f c = {};
#pragma unroll
  for (int kk = 0; kk < EE; kk += 32) {
    v16h a  = load_a16_f32(X + (size_t)r0 * EE + kk, EE);
    v16h bm = load_a16_f32(W + (size_t)c0 * EE + kk, EE);  // B[k][n] = W[n][k]
    c = wmma16(a, bm, c);
  }

  const int lane = threadIdx.x & 31;
  const int n = lane & 15, half = lane >> 4;
#pragma unroll
  for (int j = 0; j < 8; ++j) {
    const int m  = j + 8 * half;
    const int gr = r0 + m;                 // global row in [0, B*S)
    const int b  = gr / SS, s = gr % SS;
    const int o  = c0 + n;                 // feature in [0, 256)
    const int h  = o >> 5, d = o & 31;
    const float val = c[j];
    if (z == 0) {
      qh[(((size_t)(b * HH + h)) * SS + s) * DD + d] = (_Float16)(val * NORMF);
    } else if (z == 1) {
      const float gb = val + bg[o];
      gate[(size_t)gr * EE + o] = 1.0f / (1.0f + __expf(-gb));
    } else if (z == 2) {
      kh[(((size_t)(b * HH + h)) * SS + s) * DD + d] = (_Float16)val;
    } else {
      vhT[(((size_t)(b * HH + h)) * DD + d) * SS + s] = (_Float16)val;
    }
  }
}

// =====================================================================
// Kernel 2: flash attention with pair bias + gating epilogue.
// One wave per (b, h, 16 query rows).  Keys in chunks of 32; one WMMA per
// 16x16 score tile (K=32 == D).  The 256 MB bias stream is DMA'd by the
// Tensor Data Mover into a double-buffered LDS tile, overlapped with the
// WMMA/softmax of the previous chunk (TENSORcnt pipeline).
// =====================================================================
__global__ __launch_bounds__(32)
void attn_kernel(const _Float16* __restrict__ qh, const _Float16* __restrict__ kh,
                 const _Float16* __restrict__ vhT, const float* __restrict__ bias,
                 const float* __restrict__ mask, const float* __restrict__ gate,
                 _Float16* __restrict__ oG) {
  __shared__ _Float16 ldsP[16][32];        // P transpose buffer (C -> A layout)
  __shared__ float    ldsBias[2][16][32];  // TDM double buffer (2 x 2KB)

  const int qt = blockIdx.x;   // query tile
  const int h  = blockIdx.y;
  const int b  = blockIdx.z;
  const int s0 = qt * 16;
  const size_t bh = (size_t)b * HH + h;
  const _Float16* qbase = qh  + (bh * SS + s0) * DD;
  const _Float16* kbase = kh  + bh * SS * DD;
  const _Float16* vbase = vhT + bh * DD * SS;
  const float*    brow  = bias + bh * (size_t)SS * SS + (size_t)s0 * SS;

  const int lane = threadIdx.x & 31;
  const int n = lane & 15, half = lane >> 4;
  const unsigned ldsb_off[2] = {
      (unsigned)(uintptr_t)&ldsBias[0][0][0],   // generic addr[31:0] == LDS offset
      (unsigned)(uintptr_t)&ldsBias[1][0][0]};

  v16h qa = load_a16_f16(qbase, DD);      // 16 q rows x D=32 (pre-scaled)
  v8f acc0 = {}, acc1 = {};               // O tile: d 0..15 / 16..31
  float mrow[8], lrow[8];
#pragma unroll
  for (int j = 0; j < 8; ++j) { mrow[j] = -1e30f; lrow[j] = 0.0f; }

  // prologue: kick off DMA of the first bias tile
  tdm_load_bias_tile(brow, ldsb_off[0]);

  for (int t0 = 0; t0 < SS; t0 += 32) {
    const int buf = (t0 >> 5) & 1;
    // issue DMA for the next bias tile behind this iteration's compute
    if (t0 + 32 < SS) tdm_load_bias_tile(brow + (t0 + 32), ldsb_off[buf ^ 1]);

    // ---- scores: S = q @ k^T  (two 16-key tiles, one WMMA each) ----
    v16h kb0 = load_a16_f16(kbase + (size_t)t0 * DD, DD);
    v16h kb1 = load_a16_f16(kbase + (size_t)(t0 + 16) * DD, DD);
    v8f s0v = {}, s1v = {};
    s0v = wmma16(qa, kb0, s0v);
    s1v = wmma16(qa, kb1, s1v);

    // ---- wait for this iteration's bias tile (leave next DMA in flight) ----
    if (t0 + 32 < SS) __builtin_amdgcn_s_wait_tensorcnt(1);
    else              __builtin_amdgcn_s_wait_tensorcnt(0);
    asm volatile("" ::: "memory");   // don't hoist LDS reads above the wait

    // ---- + mask + pair bias (from LDS tile) ----
    const float mk0 = mask[(size_t)b * SS + t0 + n];
    const float mk1 = mask[(size_t)b * SS + t0 + 16 + n];
#pragma unroll
    for (int j = 0; j < 8; ++j) {
      const int m = j + 8 * half;
      s0v[j] += mk0 + ldsBias[buf][m][n];
      s1v[j] += mk1 + ldsBias[buf][m][n + 16];
    }

    // ---- online softmax (rows live across 16-lane halves) ----
#pragma unroll
    for (int j = 0; j < 8; ++j) {
      const float rmax  = hmax16(fmaxf(s0v[j], s1v[j]));
      const float mnew  = fmaxf(mrow[j], rmax);
      const float alpha = __expf(mrow[j] - mnew);
      const float p0 = __expf(s0v[j] - mnew);
      const float p1 = __expf(s1v[j] - mnew);
      lrow[j] = lrow[j] * alpha + hsum16(p0 + p1);
      acc0[j] *= alpha;
      acc1[j] *= alpha;
      mrow[j] = mnew;
      const int m = j + 8 * half;
      ldsP[m][n]      = (_Float16)p0;   // transpose P through LDS
      ldsP[m][n + 16] = (_Float16)p1;
    }
    __syncthreads();

    // ---- O += P @ V (A = P 16x32 from LDS, B = vhT contiguous rows) ----
    v16h pa  = load_a16_f16(&ldsP[0][0], 32);
    v16h vb0 = load_a16_f16(vbase + t0, SS);                    // d = 0..15
    v16h vb1 = load_a16_f16(vbase + (size_t)16 * SS + t0, SS);  // d = 16..31
    acc0 = wmma16(pa, vb0, acc0);
    acc1 = wmma16(pa, vb1, acc1);
    __syncthreads();
  }

  // ---- normalize, apply sigmoid gate, store f16 for final projection ----
#pragma unroll
  for (int j = 0; j < 8; ++j) {
    const int m    = j + 8 * half;
    const int srow = s0 + m;
    const float inv = 1.0f / lrow[j];
    const size_t go = ((size_t)b * SS + srow) * EE + h * DD;
    const float g0 = gate[go + n];
    const float g1 = gate[go + 16 + n];
    oG[go + n]      = (_Float16)(acc0[j] * inv * g0);
    oG[go + 16 + n] = (_Float16)(acc1[j] * inv * g1);
  }
}

// =====================================================================
// Kernel 3: output projection  out = oG @ Wo^T + bo   (fp32 out)
// =====================================================================
__global__ __launch_bounds__(32)
void outproj_kernel(const _Float16* __restrict__ oG, const float* __restrict__ Wo,
                    const float* __restrict__ bo, float* __restrict__ out) {
  const int r0 = blockIdx.x * 16;
  const int c0 = blockIdx.y * 16;
  v8f c = {};
#pragma unroll
  for (int kk = 0; kk < EE; kk += 32) {
    v16h a  = load_a16_f16(oG + (size_t)r0 * EE + kk, EE);
    v16h bm = load_a16_f32(Wo + (size_t)c0 * EE + kk, EE);
    c = wmma16(a, bm, c);
  }
  const int lane = threadIdx.x & 31;
  const int n = lane & 15, half = lane >> 4;
#pragma unroll
  for (int j = 0; j < 8; ++j) {
    const int m = j + 8 * half;
    out[(size_t)(r0 + m) * EE + c0 + n] = c[j] + bo[c0 + n];
  }
}

// =====================================================================
extern "C" void kernel_launch(void* const* d_in, const int* in_sizes, int n_in,
                              void* d_out, int out_size, void* d_ws, size_t ws_size,
                              hipStream_t stream) {
  const float* q    = (const float*)d_in[0];
  const float* k    = (const float*)d_in[1];
  const float* v    = (const float*)d_in[2];
  const float* bias = (const float*)d_in[3];
  const float* mask = (const float*)d_in[4];
  const float* Wq   = (const float*)d_in[5];
  const float* Wk   = (const float*)d_in[6];
  const float* Wv   = (const float*)d_in[7];
  const float* Wo   = (const float*)d_in[8];
  const float* bo   = (const float*)d_in[9];
  const float* Wg   = (const float*)d_in[10];
  const float* bg   = (const float*)d_in[11];
  float* out = (float*)d_out;

  // workspace layout: qh | kh | vhT (f16, 4 MB each) | gate (f32, 8 MB) | oG (f16, 4 MB)
  char* ws = (char*)d_ws;
  const size_t nh = (size_t)BB * HH * SS * DD;      // 2M elements
  _Float16* qh  = (_Float16*)ws;  ws += nh * sizeof(_Float16);
  _Float16* kh  = (_Float16*)ws;  ws += nh * sizeof(_Float16);
  _Float16* vhT = (_Float16*)ws;  ws += nh * sizeof(_Float16);
  float*    gate = (float*)ws;    ws += (size_t)BB * SS * EE * sizeof(float);
  _Float16* oG  = (_Float16*)ws;

  dim3 g1((BB * SS) / 16, EE / 16, 4);
  proj_kernel<<<g1, 32, 0, stream>>>(q, k, v, Wq, Wk, Wv, Wg, bg, qh, kh, vhT, gate);

  dim3 g2(SS / 16, HH, BB);
  attn_kernel<<<g2, 32, 0, stream>>>(qh, kh, vhT, bias, mask, gate, oG);

  dim3 g3((BB * SS) / 16, EE / 16);
  outproj_kernel<<<g3, 32, 0, stream>>>(oG, Wo, bo, out);
}
